// Attention_83021717832049
// MI455X (gfx1250) — matrix operand order repaired and checked by reading, not compiled, not run on hardware
//
#include <hip/hip_runtime.h>
#include <hip/hip_bf16.h>

// ---------------- problem constants ----------------
constexpr int Bb = 4;
constexpr int Nn = 2048;
constexpr int Cc = 768;
constexpr int Hh = 12;
constexpr int Dd = 64;          // head dim
constexpr int OC = 3 * Cc;      // 2304
constexpr int Mtot = Bb * Nn;   // 8192 rows
constexpr float QSCALE = 0.125f; // 1/sqrt(64)

// ---------------- types ----------------
typedef __attribute__((ext_vector_type(16))) __bf16 v16bf;
typedef __attribute__((ext_vector_type(8)))  __bf16 v8bf;
typedef __attribute__((ext_vector_type(8)))  float  v8f;

union FragA { v16bf v; v8bf h[2]; };

// native f32 -> bf16 conversion (backend emits hardware cvt, packs pairs)
__device__ __forceinline__ __bf16 f2bf(float f) { return (__bf16)f; }

__device__ __forceinline__ v8f wmma_bf16(v16bf a, v16bf b, v8f c) {
    return __builtin_amdgcn_wmma_f32_16x16x32_bf16(
        /*neg_a=*/false, a, /*neg_b=*/false, b,
        /*c_mod=*/(short)0, c, /*reuse_a=*/false, /*reuse_b=*/false);
}

// Async global->LDS 16B copy (gfx1250, tracked by ASYNCcnt).
// lds_off: workgroup-relative LDS byte offset; gaddr: 64-bit global address.
__device__ __forceinline__ void async_copy_b128(unsigned lds_off, const void* gaddr) {
    asm volatile("global_load_async_to_lds_b128 %0, %1, off"
                 :: "v"(lds_off), "v"(gaddr)
                 : "memory");
}
__device__ __forceinline__ void wait_asynccnt0() {
    asm volatile("s_wait_asynccnt 0x0" ::: "memory");
}
__device__ __forceinline__ unsigned lds_off_of(const void* p) {
    return (unsigned)(uintptr_t)p;   // low 32 bits of shared aperture = LDS offset
}

// =====================================================================
// Kernel 1: qkv = x @ w_qkv^T  (8192 x 2304, K=768), scatter to Q/K/V
// Grid (OC/64, Mtot/128), block 256 (8 waves). Wave w: rows [w*16,w*16+16)
// of the 128-row tile, all 64 cols -> 4 WMMA accumulators.
// =====================================================================
__global__ __launch_bounds__(256)
void qkv_proj_kernel(const float* __restrict__ x, const float* __restrict__ w,
                     __bf16* __restrict__ Qo, __bf16* __restrict__ Ko,
                     __bf16* __restrict__ Vo)
{
    __shared__ __align__(32) __bf16 As[128 * 32];  // [row][k]
    __shared__ __align__(32) __bf16 Bs[64 * 32];   // [col][k]

    const int tid  = threadIdx.x;
    const int wave = tid >> 5;
    const int lane = tid & 31;
    const int l15  = lane & 15;
    const int mBase = blockIdx.y * 128;
    const int oBase = blockIdx.x * 64;

    v8f acc[4];
    #pragma unroll
    for (int j = 0; j < 4; ++j)
        #pragma unroll
        for (int i = 0; i < 8; ++i) acc[j][i] = 0.0f;

    const int aRow = tid >> 1;              // 0..127
    const int aC0  = (tid & 1) * 16;        // 0 or 16
    const int bRow = tid >> 2;              // 0..63
    const int bC0  = (tid & 3) * 8;         // 0,8,16,24
    const int k0A  = (lane < 16) ? 0 : 8;
    const int k0B  = (lane < 16) ? 0 : 8;

    for (int kk = 0; kk < Cc; kk += 32) {
        __syncthreads();
        // stage A: 128x32 f32 -> bf16
        {
            const float4* src = (const float4*)(x + (size_t)(mBase + aRow) * Cc + kk + aC0);
            __bf16* dst = &As[aRow * 32 + aC0];
            #pragma unroll
            for (int q = 0; q < 4; ++q) {
                float4 f = src[q];
                dst[q*4+0] = f2bf(f.x); dst[q*4+1] = f2bf(f.y);
                dst[q*4+2] = f2bf(f.z); dst[q*4+3] = f2bf(f.w);
            }
        }
        // stage B: 64x32 f32 -> bf16 (rows of w_qkv = output cols)
        {
            const float4* src = (const float4*)(w + (size_t)(oBase + bRow) * Cc + kk + bC0);
            __bf16* dst = &Bs[bRow * 32 + bC0];
            #pragma unroll
            for (int q = 0; q < 2; ++q) {
                float4 f = src[q];
                dst[q*4+0] = f2bf(f.x); dst[q*4+1] = f2bf(f.y);
                dst[q*4+2] = f2bf(f.z); dst[q*4+3] = f2bf(f.w);
            }
        }
        __syncthreads();

        FragA a;
        a.h[0] = *(const v8bf*)&As[(wave * 16 + l15) * 32 + k0A];
        a.h[1] = *(const v8bf*)&As[(wave * 16 + l15) * 32 + k0A + 16];
        #pragma unroll
        for (int j = 0; j < 4; ++j) {
            FragA bu;
            bu.h[0] = *(const v8bf*)&Bs[(j * 16 + l15) * 32 + k0B];
            bu.h[1] = *(const v8bf*)&Bs[(j * 16 + l15) * 32 + k0B + 16];
            v16bf b = bu.v;
            acc[j] = wmma_bf16(a.v, b, acc[j]);
        }
    }

    // scatter to Q/K/V (B,H,N,D), fold softmax scale into Q
    #pragma unroll
    for (int j = 0; j < 4; ++j) {
        // head/which indices are constant over the 16-column sub-tile
        int oj    = oBase + j * 16;
        int which = oj / Cc;
        int rem   = oj - which * Cc;
        int h     = rem >> 6;
        int dBase = rem & 63;
        #pragma unroll
        for (int r = 0; r < 8; ++r) {
            int row = mBase + wave * 16 + r + ((lane & 16) ? 8 : 0);
            int b   = row >> 11;            // /2048
            int n   = row & 2047;
            size_t idx = (((size_t)(b * Hh + h)) * Nn + n) * Dd + dBase + l15;
            float v = acc[j][r];
            if (which == 0)      Qo[idx] = f2bf(v * QSCALE);
            else if (which == 1) Ko[idx] = f2bf(v);
            else                 Vo[idx] = f2bf(v);
        }
    }
}

// =====================================================================
// Kernel 2: flash attention. Grid (N/128, B*H), block 256 (8 waves).
// Wave owns 16 query rows; loops over 64-key blocks with online softmax.
// K tile staged via async global->LDS copy (contiguous 8KB block).
// =====================================================================
__global__ __launch_bounds__(256)
void flash_attn_kernel(const __bf16* __restrict__ Qg, const __bf16* __restrict__ Kg,
                       const __bf16* __restrict__ Vg, __bf16* __restrict__ Og)
{
    __shared__ __align__(32) __bf16 Ks[64 * 64];      // [key][d] 8KB
    __shared__ __align__(32) __bf16 Vs[64 * 64];      // [d][key] (transposed) 8KB
    __shared__ __align__(32) __bf16 Ps[8][16 * 64];   // per-wave P tile [m][key] 16KB

    const int tid  = threadIdx.x;
    const int wave = tid >> 5;
    const int lane = tid & 31;
    const int l15  = lane & 15;
    const int hi16 = (lane & 16) ? 8 : 0;

    const int bh = blockIdx.y;                  // b*H + h
    const int b  = bh / Hh;
    const int h  = bh - b * Hh;
    const __bf16* Qp = Qg + (size_t)bh * Nn * Dd;
    const __bf16* Kp = Kg + (size_t)bh * Nn * Dd;
    const __bf16* Vp = Vg + (size_t)bh * Nn * Dd;

    const int qBase = blockIdx.x * 128;
    const int qr0   = qBase + wave * 16;

    // Q fragments (held in registers the whole kernel): kdim halves 0-31, 32-63
    const int k0A = (lane < 16) ? 0 : 8;
    FragA qf[2];
    #pragma unroll
    for (int hf = 0; hf < 2; ++hf) {
        const __bf16* base = Qp + (size_t)(qr0 + l15) * Dd + hf * 32 + k0A;
        qf[hf].h[0] = *(const v8bf*)(base);
        qf[hf].h[1] = *(const v8bf*)(base + 16);
    }

    // running softmax state: 8 rows per lane (rows r+hi16)
    float mrun[8], lsum[8];
    v8f oacc[4];
    #pragma unroll
    for (int r = 0; r < 8; ++r) { mrun[r] = -3.0e38f; lsum[r] = 0.0f; }
    #pragma unroll
    for (int j = 0; j < 4; ++j)
        #pragma unroll
        for (int i = 0; i < 8; ++i) oacc[j][i] = 0.0f;

    const int ldKey = tid >> 2;            // 0..63
    const int ldD0  = (tid & 3) * 16;      // 0,16,32,48
    const int k0B   = (lane < 16) ? 0 : 8;
    const unsigned ksBase = lds_off_of(&Ks[0]);

    for (int t = 0; t < Nn / 64; ++t) {
        const int keyBase = t * 64;
        __syncthreads();
        // stage K [key][d]: contiguous 8KB -> two async 16B copies per thread
        {
            const char* g = (const char*)(Kp + (size_t)keyBase * Dd) + tid * 16;
            unsigned l0 = ksBase + (unsigned)(tid * 16);
            async_copy_b128(l0,          g);
            async_copy_b128(l0 + 4096u,  g + 4096);
        }
        // stage V transposed [d][key] (needs per-element transpose -> VGPR path)
        {
            v16bf vv = *(const v16bf*)(Vp + (size_t)(keyBase + ldKey) * Dd + ldD0);
            #pragma unroll
            for (int i = 0; i < 16; ++i)
                Vs[(ldD0 + i) * 64 + ldKey] = vv[i];
        }
        if (t + 1 < Nn / 64) {  // prefetch next key/value blocks (global_prefetch_b8)
            __builtin_prefetch(Kp + (size_t)(keyBase + 64 + ldKey) * Dd + ldD0, 0, 3);
            __builtin_prefetch(Vp + (size_t)(keyBase + 64 + ldKey) * Dd + ldD0, 0, 3);
        }
        wait_asynccnt0();
        __syncthreads();

        // S = Q . K^T : 4 key sub-tiles x 2 kdim halves
        v8f sacc[4];
        #pragma unroll
        for (int j = 0; j < 4; ++j) {
            #pragma unroll
            for (int i = 0; i < 8; ++i) sacc[j][i] = 0.0f;
            #pragma unroll
            for (int hf = 0; hf < 2; ++hf) {
                FragA bu;
                bu.h[0] = *(const v8bf*)&Ks[(j * 16 + l15) * 64 + hf * 32 + k0B];
                bu.h[1] = *(const v8bf*)&Ks[(j * 16 + l15) * 64 + hf * 32 + k0B + 16];
                v16bf bfr = bu.v;
                sacc[j] = wmma_bf16(qf[hf].v, bfr, sacc[j]);
            }
        }

        // online softmax per row (rows live in 16-lane half-wave groups)
        #pragma unroll
        for (int r = 0; r < 8; ++r) {
            float mr = sacc[0][r];
            #pragma unroll
            for (int j = 1; j < 4; ++j) mr = fmaxf(mr, sacc[j][r]);
            #pragma unroll
            for (int off = 1; off < 16; off <<= 1)
                mr = fmaxf(mr, __shfl_xor(mr, off, 32));
            float mnew  = fmaxf(mrun[r], mr);
            float alpha = __expf(mrun[r] - mnew);
            mrun[r] = mnew;
            float rs = 0.0f;
            #pragma unroll
            for (int j = 0; j < 4; ++j) {
                float p = __expf(sacc[j][r] - mnew);
                sacc[j][r] = p;
                rs += p;
            }
            #pragma unroll
            for (int off = 1; off < 16; off <<= 1)
                rs += __shfl_xor(rs, off, 32);
            lsum[r] = lsum[r] * alpha + rs;
            #pragma unroll
            for (int j = 0; j < 4; ++j) oacc[j][r] *= alpha;
        }

        // write P (bf16) to per-wave LDS [m][key] to re-layout C->A fragment
        #pragma unroll
        for (int j = 0; j < 4; ++j)
            #pragma unroll
            for (int r = 0; r < 8; ++r)
                Ps[wave][(r + hi16) * 64 + j * 16 + l15] = f2bf(sacc[j][r]);

        FragA pf[2];
        #pragma unroll
        for (int hf = 0; hf < 2; ++hf) {
            const __bf16* base = &Ps[wave][l15 * 64 + hf * 32 + k0A];
            pf[hf].h[0] = *(const v8bf*)(base);
            pf[hf].h[1] = *(const v8bf*)(base + 16);
        }

        // O += P . V
        #pragma unroll
        for (int jd = 0; jd < 4; ++jd) {
            #pragma unroll
            for (int hf = 0; hf < 2; ++hf) {
                FragA bu;
                bu.h[0] = *(const v8bf*)&Vs[(jd * 16 + l15) * 64 + hf * 32 + k0B];
                bu.h[1] = *(const v8bf*)&Vs[(jd * 16 + l15) * 64 + hf * 32 + k0B + 16];
                v16bf bfr = bu.v;
                oacc[jd] = wmma_bf16(pf[hf].v, bfr, oacc[jd]);
            }
        }
    }

    // normalize and write O to (B, N, C) bf16
    #pragma unroll
    for (int jd = 0; jd < 4; ++jd) {
        #pragma unroll
        for (int r = 0; r < 8; ++r) {
            int n = qr0 + r + hi16;
            int c = h * Dd + jd * 16 + l15;
            float ov = oacc[jd][r] / lsum[r];
            Og[((size_t)b * Nn + n) * Cc + c] = f2bf(ov);
        }
    }
}

// =====================================================================
// Kernel 3: out = O @ w_proj^T + b  (8192 x 768, K=768), f32 output.
// A tile (already bf16) staged via async global->LDS copies.
// =====================================================================
__global__ __launch_bounds__(256)
void out_proj_kernel(const __bf16* __restrict__ Og, const float* __restrict__ w,
                     const float* __restrict__ bias, float* __restrict__ out)
{
    __shared__ __align__(32) __bf16 As[128 * 32];
    __shared__ __align__(32) __bf16 Bs[64 * 32];

    const int tid  = threadIdx.x;
    const int wave = tid >> 5;
    const int lane = tid & 31;
    const int l15  = lane & 15;
    const int mBase = blockIdx.y * 128;
    const int oBase = blockIdx.x * 64;

    v8f acc[4];
    #pragma unroll
    for (int j = 0; j < 4; ++j)
        #pragma unroll
        for (int i = 0; i < 8; ++i) acc[j][i] = 0.0f;

    const int aRow = tid >> 1;
    const int aC0  = (tid & 1) * 16;
    const int bRow = tid >> 2;
    const int bC0  = (tid & 3) * 8;
    const int k0A  = (lane < 16) ? 0 : 8;
    const int k0B  = (lane < 16) ? 0 : 8;
    const unsigned asOff = lds_off_of(&As[aRow * 32 + aC0]);

    for (int kk = 0; kk < Cc; kk += 32) {
        __syncthreads();
        {   // A already bf16: 32B per thread -> two async 16B global->LDS copies
            const char* g = (const char*)(Og + (size_t)(mBase + aRow) * Cc + kk + aC0);
            async_copy_b128(asOff,       g);
            async_copy_b128(asOff + 16u, g + 16);
        }
        {   // B: f32 -> bf16 through VGPRs
            const float4* src = (const float4*)(w + (size_t)(oBase + bRow) * Cc + kk + bC0);
            __bf16* dst = &Bs[bRow * 32 + bC0];
            #pragma unroll
            for (int q = 0; q < 2; ++q) {
                float4 f = src[q];
                dst[q*4+0] = f2bf(f.x); dst[q*4+1] = f2bf(f.y);
                dst[q*4+2] = f2bf(f.z); dst[q*4+3] = f2bf(f.w);
            }
        }
        wait_asynccnt0();
        __syncthreads();

        FragA a;
        a.h[0] = *(const v8bf*)&As[(wave * 16 + l15) * 32 + k0A];
        a.h[1] = *(const v8bf*)&As[(wave * 16 + l15) * 32 + k0A + 16];
        #pragma unroll
        for (int j = 0; j < 4; ++j) {
            FragA bu;
            bu.h[0] = *(const v8bf*)&Bs[(j * 16 + l15) * 32 + k0B];
            bu.h[1] = *(const v8bf*)&Bs[(j * 16 + l15) * 32 + k0B + 16];
            v16bf bfr = bu.v;
            acc[j] = wmma_bf16(a.v, bfr, acc[j]);
        }
    }

    #pragma unroll
    for (int j = 0; j < 4; ++j) {
        #pragma unroll
        for (int r = 0; r < 8; ++r) {
            int row = mBase + wave * 16 + r + ((lane & 16) ? 8 : 0);
            int o   = oBase + j * 16 + l15;
            out[(size_t)row * Cc + o] = acc[j][r] + bias[o];
        }
    }
}

// =====================================================================
extern "C" void kernel_launch(void* const* d_in, const int* in_sizes, int n_in,
                              void* d_out, int out_size, void* d_ws, size_t ws_size,
                              hipStream_t stream) {
    const float* x      = (const float*)d_in[0];
    const float* w_qkv  = (const float*)d_in[1];
    const float* w_proj = (const float*)d_in[2];
    const float* b_proj = (const float*)d_in[3];
    float* out = (float*)d_out;

    const size_t perBuf = (size_t)Bb * Hh * Nn * Dd;   // 6,291,456 bf16 elems
    __bf16* Qw = (__bf16*)d_ws;
    __bf16* Kw = Qw + perBuf;
    __bf16* Vw = Kw + perBuf;
    __bf16* Ow = Vw + perBuf;                           // (B,N,C) bf16

    dim3 blk(256);
    qkv_proj_kernel<<<dim3(OC / 64, Mtot / 128), blk, 0, stream>>>(x, w_qkv, Qw, Kw, Vw);
    flash_attn_kernel<<<dim3(Nn / 128, Bb * Hh), blk, 0, stream>>>(Qw, Kw, Vw, Ow);
    out_proj_kernel<<<dim3(Cc / 64, Mtot / 128), blk, 0, stream>>>(Ow, w_proj, b_proj, out);
}
